// NearestNeighbour_42820823941298
// MI455X (gfx1250) — matrix operand order repaired and checked by reading, not compiled, not run on hardware
//
#include <hip/hip_runtime.h>

typedef __attribute__((ext_vector_type(16))) __bf16 v16bf;
typedef __attribute__((ext_vector_type(4)))  __bf16 v4bf;
typedef __attribute__((ext_vector_type(8)))  float  v8f;
typedef __attribute__((ext_vector_type(4)))  unsigned int v4u;

union BfPack { v16bf v; v4u u[2]; };

#define TILES        2                    // 16-row A tiles per wave (32 rows/wave)
#define WAVES        4                    // waves per workgroup
#define ROWS_PER_WG  (WAVES * TILES * 16) // 128 A rows per workgroup
#define TILE_COLS    32                   // B rows (columns of score) per stage
#define TILE_BYTES   (TILE_COLS * 128)    // 32 * 64 * 2B = 4096

// ---------------------------------------------------------------------------
// Prep: elementwise f32 -> bf16 (A), vectorized 4 at a time
// ---------------------------------------------------------------------------
__global__ void convert_bf16_kernel(const float* __restrict__ in,
                                    __bf16* __restrict__ out, int n4) {
    int i = blockIdx.x * blockDim.x + threadIdx.x;
    if (i < n4) {
        float4 f = ((const float4*)in)[i];
        v4bf o = { (__bf16)f.x, (__bf16)f.y, (__bf16)f.z, (__bf16)f.w };
        ((v4bf*)out)[i] = o;
    }
}

// ---------------------------------------------------------------------------
// Prep: B f32 -> bf16 + rbn[row] = -0.5 * ||b_row||^2 (bf16-rounded values).
// One 64-thread block per B row (D = 64). Negated so it can seed the WMMA
// accumulator directly: score = a.b - 0.5||b||^2, argmax == 1-NN.
// ---------------------------------------------------------------------------
__global__ void prep_b_kernel(const float* __restrict__ B,
                              __bf16* __restrict__ Bb,
                              float* __restrict__ rbn) {
    __shared__ float red[64];
    int row = blockIdx.x, k = threadIdx.x;
    float v  = B[row * 64 + k];
    __bf16 b = (__bf16)v;
    Bb[row * 64 + k] = b;
    float vb = (float)b;
    red[k] = vb * vb;
    __syncthreads();
    #pragma unroll
    for (int off = 32; off > 0; off >>= 1) {
        if (k < off) red[k] += red[k + off];
        __syncthreads();
    }
    if (k == 0) rbn[row] = -0.5f * red[0];
}

// ---------------------------------------------------------------------------
// Main kernel
// ---------------------------------------------------------------------------
__global__ __launch_bounds__(WAVES * 32) void nn_wmma_kernel(
        const __bf16* __restrict__ Ab, const __bf16* __restrict__ Bb,
        const float* __restrict__ rbn, int* __restrict__ out, int steps) {
    __shared__ alignas(16) char smem[2][TILE_BYTES];

    const int tid  = threadIdx.x;
    const int lane = tid & 31;
    const int wave = tid >> 5;
    const int lo   = lane & 15;   // A row / B column within 16-tile
    const int hi   = lane >> 4;   // K-chunk half held by this lane
    const int rowBase = blockIdx.x * ROWS_PER_WG + wave * (TILES * 16);

    // ---- loop-invariant A operands (bf16 16x32 layout, D=64 -> 2 K-chunks) -
    BfPack a0[TILES], a1[TILES];
    #pragma unroll
    for (int t = 0; t < TILES; ++t) {
        const char* ar = (const char*)Ab + (size_t)(rowBase + t * 16 + lo) * 128;
        a0[t].u[0] = *(const v4u*)(ar +      hi * 16);  // K 0..7   / 8..15
        a0[t].u[1] = *(const v4u*)(ar + 32 + hi * 16);  // K 16..23 / 24..31
        a1[t].u[0] = *(const v4u*)(ar + 64 + hi * 16);  // K 32..39 / 40..47
        a1[t].u[1] = *(const v4u*)(ar + 96 + hi * 16);  // K 48..55 / 56..63
    }

    float best[TILES][8];
    int   bidx[TILES][8];
    #pragma unroll
    for (int t = 0; t < TILES; ++t)
        #pragma unroll
        for (int s = 0; s < 8; ++s) { best[t][s] = -3.0e38f; bidx[t][s] = 0; }

    // one async b128 pair per thread stages a full 4 KB tile per workgroup
    auto stage = [&](int tile, char* dst) {
        unsigned ldst = (unsigned)(size_t)(dst + tid * 32);
        unsigned goff = (unsigned)(tile * TILE_BYTES + tid * 32);
        asm volatile("global_load_async_to_lds_b128 %0, %1, %2"
                     :: "v"(ldst), "v"(goff), "s"((const void*)Bb) : "memory");
        asm volatile("global_load_async_to_lds_b128 %0, %1, %2 offset:16"
                     :: "v"(ldst), "v"(goff), "s"((const void*)Bb) : "memory");
    };

    // 8 WMMAs (2 column sub-tiles x 2 A tiles x 2 K-chunks) per staged tile;
    // accumulator seeds (-0.5||b||^2) arrive pre-loaded one tile ahead.
    auto compute = [&](const char* tb, int j2, float nrb0, float nrb1) {
        #pragma unroll
        for (int c = 0; c < 2; ++c) {
            const char* brow = tb + (c * 16 + lo) * 128 + hi * 16;
            BfPack b0, b1;
            b0.u[0] = *(const v4u*)(brow +  0);
            b0.u[1] = *(const v4u*)(brow + 32);
            b1.u[0] = *(const v4u*)(brow + 64);
            b1.u[1] = *(const v4u*)(brow + 96);
            const int col = j2 * TILE_COLS + c * 16 + lo;
            const float nrb = (c == 0) ? nrb0 : nrb1;
            #pragma unroll
            for (int t = 0; t < TILES; ++t) {
                v8f acc;
                #pragma unroll
                for (int s = 0; s < 8; ++s) acc[s] = nrb;
                acc = __builtin_amdgcn_wmma_f32_16x16x32_bf16(
                        false, a0[t].v, false, b0.v, (short)0, acc, false, false);
                acc = __builtin_amdgcn_wmma_f32_16x16x32_bf16(
                        false, a1[t].v, false, b1.v, (short)0, acc, false, false);
                #pragma unroll
                for (int s = 0; s < 8; ++s) {
                    float sc = acc[s];
                    if (sc > best[t][s]) { best[t][s] = sc; bidx[t][s] = col; }
                }
            }
        }
    };

    const int last = steps - 1;

    // prologue: tile 0 -> buffer 0; seed values for tile 0
    stage(0, smem[0]);
    float r0a = rbn[lo];
    float r0b = rbn[16 + lo];
    asm volatile("s_wait_asynccnt 0x0" ::: "memory");
    __syncthreads();

    // software-pipelined, statically double-buffered (steps is even: M/32).
    // B tiles AND rbn seeds are both prefetched one tile ahead.
    for (int jj = 0; jj < steps; jj += 2) {
        const int t1 = min(jj + 1, last);
        stage(t1, smem[1]);
        float r1a = rbn[t1 * TILE_COLS + lo];
        float r1b = rbn[t1 * TILE_COLS + 16 + lo];
        compute(smem[0], jj, r0a, r0b);
        asm volatile("s_wait_asynccnt 0x0" ::: "memory");
        __syncthreads();

        const int t2 = min(jj + 2, last);
        stage(t2, smem[0]);
        r0a = rbn[t2 * TILE_COLS + lo];
        r0b = rbn[t2 * TILE_COLS + 16 + lo];
        compute(smem[1], jj + 1, r1a, r1b);
        asm volatile("s_wait_asynccnt 0x0" ::: "memory");
        __syncthreads();
    }

    // ---- cross-lane argmax over the 16 lanes holding each output row -------
    #pragma unroll
    for (int t = 0; t < TILES; ++t) {
        #pragma unroll
        for (int s = 0; s < 8; ++s) {
            float v = best[t][s];
            int  id = bidx[t][s];
            #pragma unroll
            for (int m = 1; m <= 8; m <<= 1) {    // stays within 16-lane group
                float ov = __shfl_xor(v, m, 32);
                int   oi = __shfl_xor(id, m, 32);
                if (ov > v || (ov == v && oi < id)) { v = ov; id = oi; }
            }
            if (lo == 0)
                out[rowBase + t * 16 + s + hi * 8] = id;
        }
    }
}

// ---------------------------------------------------------------------------
extern "C" void kernel_launch(void* const* d_in, const int* in_sizes, int n_in,
                              void* d_out, int out_size, void* d_ws, size_t ws_size,
                              hipStream_t stream) {
    const float* A = (const float*)d_in[0];
    const float* B = (const float*)d_in[1];
    const int D = 64;
    const int N = in_sizes[0] / D;   // 32768
    const int M = in_sizes[1] / D;   // 16384

    // workspace layout: A_bf16 (N*D*2) | B_bf16 (M*D*2) | rbn (M*4)
    char*  ws  = (char*)d_ws;
    __bf16* Ab = (__bf16*)ws;
    __bf16* Bb = (__bf16*)(ws + (size_t)N * D * 2);
    float* rbn = (float*)(ws + (size_t)(N + M) * D * 2);

    int n4 = (N * D) / 4;
    convert_bf16_kernel<<<(n4 + 255) / 256, 256, 0, stream>>>(A, Ab, n4);
    prep_b_kernel<<<M, 64, 0, stream>>>(B, Bb, rbn);
    nn_wmma_kernel<<<N / ROWS_PER_WG, WAVES * 32, 0, stream>>>(
        Ab, Bb, rbn, (int*)d_out, M / TILE_COLS);
}
